// BitterGAT_Baseline_52475910422829
// MI455X (gfx1250) — matrix operand order repaired and checked by reading, not compiled, not run on hardware
//
#include <hip/hip_runtime.h>
#include <hip/hip_bf16.h>

// Problem constants (from reference)
#define N_NODES 50000
#define N_EDGES 800000
#define N_GRAPH 256
#define IN_DIM  20
#define HID     64
#define OUT_DIM 2
#define TOT_E   (N_EDGES + N_NODES)   // edges + self loops

typedef __bf16 bf16_t;
typedef bf16_t v16bf __attribute__((ext_vector_type(16)));
typedef float  v8f   __attribute__((ext_vector_type(8)));

struct B32x2 { uint4 a, b; };

__device__ inline v16bf load_frag(const bf16_t* p0, const bf16_t* p1) {
  B32x2 u;
  u.a = *reinterpret_cast<const uint4*>(p0);
  u.b = *reinterpret_cast<const uint4*>(p1);
  return __builtin_bit_cast(v16bf, u);
}

// ---- order-preserving float <-> uint mapping for atomic max ----
__device__ inline unsigned f2o(float f) {
  unsigned u = __float_as_uint(f);
  return (u & 0x80000000u) ? ~u : (u | 0x80000000u);
}
__device__ inline float o2f(unsigned k) {
  unsigned u = (k & 0x80000000u) ? (k & 0x7fffffffu) : ~k;
  return __uint_as_float(u);
}

// ---------------- conversion kernels (f32 -> bf16 hi/lo split) ----------------
__global__ void k_convert_A(const float* __restrict__ X, bf16_t* __restrict__ hi,
                            bf16_t* __restrict__ lo, int n, int Kin, int Kpad) {
  int idx = blockIdx.x * blockDim.x + threadIdx.x;
  if (idx >= n * Kpad) return;
  int row = idx / Kpad, k = idx % Kpad;
  float v = (k < Kin) ? X[row * Kin + k] : 0.0f;
  bf16_t h = (bf16_t)v;
  hi[idx] = h;
  lo[idx] = (bf16_t)(v - (float)h);
}

// transpose W[K][64] into column-major Wt[n][Kpad] (hi/lo split, zero padded)
__global__ void k_convert_W(const float* __restrict__ W, bf16_t* __restrict__ hi,
                            bf16_t* __restrict__ lo, int Kin, int Kpad) {
  int idx = blockIdx.x * blockDim.x + threadIdx.x;
  if (idx >= HID * Kpad) return;
  int n = idx / Kpad, k = idx % Kpad;
  float v = (k < Kin) ? W[k * HID + n] : 0.0f;
  bf16_t h = (bf16_t)v;
  hi[idx] = h;
  lo[idx] = (bf16_t)(v - (float)h);
}

// ---------------- WMMA GEMM: H[n x 64] = A[n x K] * W[K x 64] -----------------
// One wave computes a 16x64 strip using v_wmma_f32_16x16x32_bf16 with
// bf16 hi/lo split accumulation (near-f32 accuracy, f32 accumulator).
__global__ void k_gemm_wmma(const bf16_t* __restrict__ Ahi, const bf16_t* __restrict__ Alo,
                            const bf16_t* __restrict__ Bhi, const bf16_t* __restrict__ Blo,
                            float* __restrict__ H, int nrows, int Kpad) {
  const int lane = threadIdx.x & 31;
  const int wave = threadIdx.x >> 5;
  const int tile = blockIdx.x * (blockDim.x >> 5) + wave;
  if (tile * 16 >= nrows) return;            // wave-uniform guard: EXEC stays all-ones
  const int row0 = tile * 16;
  const int r  = lane & 15;                  // row (A) / col (B) within tile
  const int hh = lane >> 4;                  // lane half selects K sub-block

  v8f acc[4] = {v8f{}, v8f{}, v8f{}, v8f{}};
  for (int k0 = 0; k0 < Kpad; k0 += 32) {
    const bf16_t* pah = Ahi + (size_t)(row0 + r) * Kpad + k0 + 8 * hh;
    const bf16_t* pal = Alo + (size_t)(row0 + r) * Kpad + k0 + 8 * hh;
    v16bf a_hi = load_frag(pah, pah + 16);   // K = {k0+8h .. +7} U {k0+8h+16 .. +7}
    v16bf a_lo = load_frag(pal, pal + 16);
#pragma unroll
    for (int t = 0; t < 4; ++t) {
      const bf16_t* pbh = Bhi + (size_t)(t * 16 + r) * Kpad + k0 + 8 * hh;
      const bf16_t* pbl = Blo + (size_t)(t * 16 + r) * Kpad + k0 + 8 * hh;
      v16bf b_hi = load_frag(pbh, pbh + 16); // K = {k0+8h .. +7} U {k0+8h+16 .. +7}
      v16bf b_lo = load_frag(pbl, pbl + 16);
      acc[t] = __builtin_amdgcn_wmma_f32_16x16x32_bf16(false, a_hi, false, b_hi,
                                                       (short)0, acc[t], false, false);
      acc[t] = __builtin_amdgcn_wmma_f32_16x16x32_bf16(false, a_hi, false, b_lo,
                                                       (short)0, acc[t], false, false);
      acc[t] = __builtin_amdgcn_wmma_f32_16x16x32_bf16(false, a_lo, false, b_hi,
                                                       (short)0, acc[t], false, false);
    }
  }
#pragma unroll
  for (int t = 0; t < 4; ++t)
#pragma unroll
    for (int j = 0; j < 8; ++j)
      H[(size_t)(row0 + j + 8 * hh) * HID + t * 16 + r] = acc[t][j];
}

// ---------------- attention scalars: al_s = H a_src ; al_d = H a_dst ----------
__global__ void k_al(const float* __restrict__ H, const float* __restrict__ a_s,
                     const float* __restrict__ a_d, float* __restrict__ als,
                     float* __restrict__ ald, int n) {
  int i = blockIdx.x * blockDim.x + threadIdx.x;
  if (i >= n) return;
  float s = 0.f, d = 0.f;
#pragma unroll 8
  for (int f = 0; f < HID; ++f) {
    float h = H[(size_t)i * HID + f];
    s += h * a_s[f];
    d += h * a_d[f];
  }
  als[i] = s;
  ald[i] = d;
}

// ---------------- per-layer init: mord=0 (== -inf key floor), den=0, hout=0 ---
__global__ void k_init(unsigned* __restrict__ mord, float* __restrict__ den,
                       float* __restrict__ hout, int n) {
  int idx = blockIdx.x * blockDim.x + threadIdx.x;
  if (idx < n) { mord[idx] = 0u; den[idx] = 0.f; }
  if (idx < n * HID) hout[idx] = 0.f;
}

// ---------------- edge pass 1: e = leaky(al_s[src]+al_d[dst]); segment max ----
__global__ void k_edge_max(const int* __restrict__ ei, const float* __restrict__ als,
                           const float* __restrict__ ald, float* __restrict__ ev,
                           unsigned* __restrict__ mord) {
  int e = blockIdx.x * blockDim.x + threadIdx.x;
  if (e >= TOT_E) return;
  int s, d;
  if (e < N_EDGES) { s = ei[e]; d = ei[N_EDGES + e]; }
  else             { s = d = e - N_EDGES; }
  float v = als[s] + ald[d];
  v = (v > 0.f) ? v : 0.2f * v;          // LeakyReLU(0.2)
  ev[e] = v;
  atomicMax(&mord[d], f2o(v));
}

// ---------------- edge pass 2: ex = exp(e - m[dst]); segment sum --------------
__global__ void k_edge_exp(const int* __restrict__ ei, const float* __restrict__ ev,
                           const unsigned* __restrict__ mord, float* __restrict__ den,
                           float* __restrict__ exv) {
  int e = blockIdx.x * blockDim.x + threadIdx.x;
  if (e >= TOT_E) return;
  int d = (e < N_EDGES) ? ei[N_EDGES + e] : e - N_EDGES;
  float m = o2f(mord[d]);
  float ex = expf(ev[e] - m);
  exv[e] = ex;
  atomicAdd(&den[d], ex);
}

// ---------------- edge pass 3: out[dst] += alpha * h[src] ---------------------
__global__ void k_edge_agg(const int* __restrict__ ei, const float* __restrict__ H,
                           const float* __restrict__ exv, const float* __restrict__ den,
                           float* __restrict__ hout) {
  int idx = blockIdx.x * blockDim.x + threadIdx.x;
  int e = idx >> 6, f = idx & 63;
  if (e >= TOT_E) return;
  int s, d;
  if (e < N_EDGES) { s = ei[e]; d = ei[N_EDGES + e]; }
  else             { s = d = e - N_EDGES; }
  float alpha = exv[e] / den[d];
  atomicAdd(&hout[(size_t)d * HID + f], alpha * H[(size_t)s * HID + f]);
}

// ---------------- bias + optional relu ----------------------------------------
__global__ void k_bias_act(float* __restrict__ H, const float* __restrict__ b,
                           int n, int relu) {
  int idx = blockIdx.x * blockDim.x + threadIdx.x;
  if (idx >= n * HID) return;
  float v = H[idx] + b[idx & 63];
  if (relu) v = fmaxf(v, 0.f);
  H[idx] = v;
}

// ---------------- pooling -----------------------------------------------------
__global__ void k_zero(float* __restrict__ p, int n) {
  int idx = blockIdx.x * blockDim.x + threadIdx.x;
  if (idx < n) p[idx] = 0.f;
}

__global__ void k_pool(const float* __restrict__ H, const int* __restrict__ batch,
                       float* __restrict__ sums, float* __restrict__ cnt, int n) {
  int idx = blockIdx.x * blockDim.x + threadIdx.x;
  if (idx >= n * HID) return;
  int i = idx >> 6, f = idx & 63;
  int g = batch[i];
  atomicAdd(&sums[(size_t)g * HID + f], H[idx]);
  if (f == 0) atomicAdd(&cnt[g], 1.0f);
}

__global__ void k_final(const float* __restrict__ sums, const float* __restrict__ cnt,
                        const float* __restrict__ linW, const float* __restrict__ linb,
                        float* __restrict__ out) {
  int idx = blockIdx.x * blockDim.x + threadIdx.x;
  if (idx >= N_GRAPH * OUT_DIM) return;
  int g = idx >> 1, o = idx & 1;
  float c = fmaxf(cnt[g], 1.0f);
  float acc = 0.f;
#pragma unroll 8
  for (int f = 0; f < HID; ++f)
    acc += (sums[(size_t)g * HID + f] / c) * linW[f * OUT_DIM + o];
  out[idx] = acc + linb[o];
}

// ---------------- host side ---------------------------------------------------
static inline int cdiv(int a, int b) { return (a + b - 1) / b; }

extern "C" void kernel_launch(void* const* d_in, const int* in_sizes, int n_in,
                              void* d_out, int out_size, void* d_ws, size_t ws_size,
                              hipStream_t stream) {
  (void)in_sizes; (void)n_in; (void)out_size; (void)ws_size;

  // Inputs in setup_inputs() order
  const float* x      = (const float*)d_in[0];
  const int*   ei     = (const int*)d_in[1];     // [2, E] flat: row0 = src, row1 = dst
  const int*   batch  = (const int*)d_in[2];
  const float* W1     = (const float*)d_in[3];
  const float* as1    = (const float*)d_in[4];
  const float* ad1    = (const float*)d_in[5];
  const float* b1     = (const float*)d_in[6];
  const float* W2     = (const float*)d_in[7];
  const float* as2    = (const float*)d_in[8];
  const float* ad2    = (const float*)d_in[9];
  const float* b2     = (const float*)d_in[10];
  const float* W3     = (const float*)d_in[11];
  const float* as3    = (const float*)d_in[12];
  const float* ad3    = (const float*)d_in[13];
  const float* b3     = (const float*)d_in[14];
  const float* linW   = (const float*)d_in[15];
  const float* linb   = (const float*)d_in[16];
  float* out = (float*)d_out;

  // Workspace bump allocator (256B aligned)
  char* base = (char*)d_ws;
  size_t off = 0;
  auto alloc = [&](size_t bytes) -> void* {
    void* p = base + off;
    off = (off + bytes + 255) & ~(size_t)255;
    return p;
  };
  float*    bufA = (float*)alloc((size_t)N_NODES * HID * sizeof(float)); // h (features)
  float*    bufB = (float*)alloc((size_t)N_NODES * HID * sizeof(float)); // aggregated out
  bf16_t*   Ahi  = (bf16_t*)alloc((size_t)N_NODES * HID * sizeof(bf16_t));
  bf16_t*   Alo  = (bf16_t*)alloc((size_t)N_NODES * HID * sizeof(bf16_t));
  bf16_t*   Whi  = (bf16_t*)alloc((size_t)HID * HID * sizeof(bf16_t));
  bf16_t*   Wlo  = (bf16_t*)alloc((size_t)HID * HID * sizeof(bf16_t));
  float*    als  = (float*)alloc((size_t)N_NODES * sizeof(float));
  float*    ald  = (float*)alloc((size_t)N_NODES * sizeof(float));
  unsigned* mord = (unsigned*)alloc((size_t)N_NODES * sizeof(unsigned));
  float*    den  = (float*)alloc((size_t)N_NODES * sizeof(float));
  float*    ev   = (float*)alloc((size_t)TOT_E * sizeof(float));
  float*    exv  = (float*)alloc((size_t)TOT_E * sizeof(float));
  float*    sums = (float*)alloc((size_t)N_GRAPH * HID * sizeof(float));
  float*    cnt  = (float*)alloc((size_t)N_GRAPH * sizeof(float));

  const int B = 256;
  const int gemm_grid = cdiv(N_NODES / 16, B / 32);   // 8 row-tiles (waves) per block

  auto layer = [&](const float* X, int Kin, const float* W, const float* a_s,
                   const float* a_d, const float* b, int relu) {
    int Kpad = (Kin + 31) & ~31;
    k_convert_A<<<cdiv(N_NODES * Kpad, B), B, 0, stream>>>(X, Ahi, Alo, N_NODES, Kin, Kpad);
    k_convert_W<<<cdiv(HID * Kpad, B), B, 0, stream>>>(W, Whi, Wlo, Kin, Kpad);
    k_gemm_wmma<<<gemm_grid, B, 0, stream>>>(Ahi, Alo, Whi, Wlo, bufA, N_NODES, Kpad);
    k_al<<<cdiv(N_NODES, B), B, 0, stream>>>(bufA, a_s, a_d, als, ald, N_NODES);
    k_init<<<cdiv(N_NODES * HID, B), B, 0, stream>>>(mord, den, bufB, N_NODES);
    k_edge_max<<<cdiv(TOT_E, B), B, 0, stream>>>(ei, als, ald, ev, mord);
    k_edge_exp<<<cdiv(TOT_E, B), B, 0, stream>>>(ei, ev, mord, den, exv);
    k_edge_agg<<<cdiv(TOT_E * HID, B), B, 0, stream>>>(ei, bufA, exv, den, bufB);
    k_bias_act<<<cdiv(N_NODES * HID, B), B, 0, stream>>>(bufB, b, N_NODES, relu);
  };

  layer(x,    IN_DIM, W1, as1, ad1, b1, 1);
  layer(bufB, HID,    W2, as2, ad2, b2, 1);
  layer(bufB, HID,    W3, as3, ad3, b3, 0);

  k_zero<<<cdiv(N_GRAPH * HID, B), B, 0, stream>>>(sums, N_GRAPH * HID);
  k_zero<<<cdiv(N_GRAPH, B), B, 0, stream>>>(cnt, N_GRAPH);
  k_pool<<<cdiv(N_NODES * HID, B), B, 0, stream>>>(bufB, batch, sums, cnt, N_NODES);
  k_final<<<cdiv(N_GRAPH * OUT_DIM, B), B, 0, stream>>>(sums, cnt, linW, linb, out);
}
